// M2DBlock_42056319762571
// MI455X (gfx1250) — matrix operand order repaired and checked by reading, not compiled, not run on hardware
//
#include <hip/hip_runtime.h>
#include <math.h>

// Problem constants (from the reference): B=2, H=W=32, DM=192, DS=16, RK=12, DI=384
#define NTOK 2048   // B*H*W tokens
#define DM_  192
#define DI_  384
#define DS_  16
#define RK_  12
#define PC_  56     // 2*RK + 2*DS columns of dbc

typedef __attribute__((ext_vector_type(16))) _Float16 v16h;
typedef __attribute__((ext_vector_type(8)))  float    v8f;

__device__ __forceinline__ float gelu_exact(float x) {
    return 0.5f * x * (1.0f + erff(x * 0.70710678118654752440f));
}
__device__ __forceinline__ float softplus_stable(float x) {
    return fmaxf(x, 0.0f) + log1pf(expf(-fabsf(x)));
}
__device__ __forceinline__ void pack4(v16h& a, int base, float4 f) {
    a[base + 0] = (_Float16)f.x;
    a[base + 1] = (_Float16)f.y;
    a[base + 2] = (_Float16)f.z;
    a[base + 3] = (_Float16)f.w;
}

// Fused GEMM + epilogue: C(M,NMAX) = A(M,KMAX) * B(KMAX,NMAX) + bias, then MODE.
// All strides/shapes are template constants so pad-masking constant-folds and
// B loads become saddr+immediate-offset (no per-load 64-bit induction).
// One wave32 per 16x16 tile, 4 waves (n-tiles) per 128-thread block.
// MODE: 0=bias, 1=bias+GELU(exact erf), 2=bias+softplus.
template<int LDA, int LDB, int LDC, int K, int KMAX, int NMAX, int MODE>
__global__ __launch_bounds__(128)
void gemm_epi_kernel(const float* __restrict__ A,
                     const float* __restrict__ Bp,
                     const float* __restrict__ bias,
                     float* __restrict__ C)
{
    constexpr bool FULLK = (K == KMAX);
    constexpr bool FULLN = (NMAX % 16 == 0);

    const int wave  = threadIdx.x >> 5;
    const int ntile = blockIdx.y * 4 + wave;
    const int n0    = ntile * 16;
    const int m0    = blockIdx.x * 16;
    const int lane  = threadIdx.x & 31;
    const int mr    = lane & 15;
    const bool hi   = (lane & 16) != 0;

    // nvalid >= 1 for every launched tile, so no EXEC-narrowing before WMMA.
    const int  nvalid = FULLN ? 16 : ((NMAX - n0) < 16 ? (NMAX - n0) : 16);
    const bool nok    = FULLN ? true : (mr < nvalid);
    const int  ncl    = FULLN ? mr : (nok ? mr : 0);      // clamped (in-bounds) column

    const float* __restrict__ Arow = A + (size_t)(m0 + mr) * LDA;
    const float* __restrict__ Bcol = Bp + n0 + ncl;

    v8f acc = {};
    #pragma unroll 1
    for (int k0 = 0; k0 < K; k0 += 32) {
        __builtin_prefetch(Bcol + (size_t)(k0 + 32) * LDB, 0, 1); // speculative
        // A fragment (16-bit A 16x32, ISA 7.12.2):
        // lanes 0-15: M=lane, K={0..7,16..23}+k0 ; lanes 16-31: M=lane-16, K={8..15,24..31}+k0
        const int ka0 = k0 + (hi ? 8 : 0);
        v16h a;
        if constexpr (FULLK) {
            pack4(a, 0,  *(const float4*)(Arow + ka0));
            pack4(a, 4,  *(const float4*)(Arow + ka0 + 4));
            pack4(a, 8,  *(const float4*)(Arow + ka0 + 16));
            pack4(a, 12, *(const float4*)(Arow + ka0 + 20));
        } else {
            #pragma unroll
            for (int t = 0; t < 8; ++t) {
                const int k1 = ka0 + t;
                const int k2 = ka0 + 16 + t;
                const float f1 = Arow[(k1 < KMAX) ? k1 : 0];   // unconditional load
                const float f2 = Arow[(k2 < KMAX) ? k2 : 0];
                a[t]     = (_Float16)((k1 < KMAX) ? f1 : 0.0f); // v_cndmask, no branch
                a[t + 8] = (_Float16)((k2 < KMAX) ? f2 : 0.0f);
            }
        }
        // B fragment (16-bit B 32x16): lanes 0-15: N=lane, K={0..7,16..23}+k0;
        // lanes 16-31: N=lane-16, K={8..15,24..31}+k0. Coalesced 64B per K row.
        const int kb0 = k0 + (hi ? 8 : 0);
        v16h b;
        #pragma unroll
        for (int t = 0; t < 16; ++t) {
            const int  k  = kb0 + (t < 8 ? t : t + 8);
            const int  kc = (FULLK || k < KMAX) ? k : 0;
            const float f = Bcol[(size_t)kc * LDB];            // immediate offset (t<8 ? t : t+8)*LDB*4
            const bool ok = nok && (FULLK || k < KMAX);
            b[t] = (_Float16)(ok ? f : 0.0f);
        }
        acc = __builtin_amdgcn_wmma_f32_16x16x32_f16(false, a, false, b,
                                                     (short)0, acc, false, false);
    }

    // C/D layout: VGPR r -> (M = r + (hi?8:0), N = lane&15)
    if (FULLN || mr < nvalid) {
        const int   n     = n0 + mr;
        const int   mbase = m0 + (hi ? 8 : 0);
        const float bv    = bias[n];
        #pragma unroll
        for (int r = 0; r < 8; ++r) {
            float v = acc[r] + bv;
            if constexpr (MODE == 1)      v = gelu_exact(v);
            else if constexpr (MODE == 2) v = softplus_stable(v);
            C[(size_t)(mbase + r) * LDC + n] = v;
        }
    }
}

// 2-D selective scan. One thread per independent (b, di, ds) channel:
// h[i,j] = 0.5*dAT*h[i-1,j] + 0.5*dAL*h[i,j-1] + 0.5*(dT+dL)*h*B.
// ds lives in the low 4 lane bits so the sum over DS=16 states is a
// 4-step shfl_xor reduction; lane ds==0 fuses +h*D and GELU.
// dAT/dAL are recomputed (2 exps) instead of materializing 150MB tensors.
__global__ __launch_bounds__(256)
void scan_kernel(const float* __restrict__ h,
                 const float* __restrict__ deltaT,
                 const float* __restrict__ deltaL,
                 const float* __restrict__ dbc,
                 const float* __restrict__ AT_log,
                 const float* __restrict__ AL_log,
                 const float* __restrict__ Dw,
                 float* __restrict__ yact)
{
    const int t  = blockIdx.x * blockDim.x + threadIdx.x; // 0..12287
    const int ds = t & 15;
    const int g  = t >> 4;          // (b, di) group, 0..767
    const int di = g % DI_;
    const int b  = g / DI_;

    const float AT = -expf(AT_log[di * DS_ + ds]);
    const float AL = -expf(AL_log[di * DS_ + ds]);
    const float Dv = Dw[di];

    float prev[32];                 // previous-row carry, in registers
    #pragma unroll
    for (int j = 0; j < 32; ++j) prev[j] = 0.0f;

    for (int i = 0; i < 32; ++i) {
        float left = 0.0f;
        #pragma unroll
        for (int j = 0; j < 32; ++j) {
            const int   tok = (b * 32 + i) * 32 + j;
            const float dT  = deltaT[(size_t)tok * DI_ + di]; // broadcast in group
            const float dL  = deltaL[(size_t)tok * DI_ + di];
            const float hv  = h[(size_t)tok * DI_ + di];
            const float Bv  = dbc[(size_t)tok * PC_ + 2 * RK_ + ds];        // Bm
            const float Cv  = dbc[(size_t)tok * PC_ + 2 * RK_ + DS_ + ds];  // Cm

            const float aT = 0.5f * expf(dT * AT);
            const float aL = 0.5f * expf(dL * AL);
            const float u  = 0.5f * (dT + dL) * hv * Bv;

            const float s = fmaf(aT, prev[j], fmaf(aL, left, u));
            prev[j] = s;
            left    = s;

            float contrib = s * Cv;
            contrib += __shfl_xor(contrib, 8, 32);
            contrib += __shfl_xor(contrib, 4, 32);
            contrib += __shfl_xor(contrib, 2, 32);
            contrib += __shfl_xor(contrib, 1, 32);
            if (ds == 0) {
                yact[(size_t)tok * DI_ + di] = gelu_exact(contrib + hv * Dv);
            }
        }
    }
}

extern "C" void kernel_launch(void* const* d_in, const int* in_sizes, int n_in,
                              void* d_out, int out_size, void* d_ws, size_t ws_size,
                              hipStream_t stream) {
    (void)in_sizes; (void)n_in; (void)out_size; (void)ws_size;
    const float* x      = (const float*)d_in[0];
    const float* W_in   = (const float*)d_in[1];
    const float* b_in   = (const float*)d_in[2];
    const float* W_x    = (const float*)d_in[3];
    const float* b_x    = (const float*)d_in[4];
    const float* W_dtT  = (const float*)d_in[5];
    const float* b_dtT  = (const float*)d_in[6];
    const float* W_dtL  = (const float*)d_in[7];
    const float* b_dtL  = (const float*)d_in[8];
    const float* AT_log = (const float*)d_in[9];
    const float* AL_log = (const float*)d_in[10];
    const float* Dw     = (const float*)d_in[11];
    const float* W_out  = (const float*)d_in[12];
    const float* b_out  = (const float*)d_in[13];
    float* out = (float*)d_out;

    // Workspace carve-up (~13 MB, resident in L2)
    float* ws     = (float*)d_ws;
    float* h      = ws;                              // NTOK*DI
    float* dT     = h    + (size_t)NTOK * DI_;       // NTOK*DI
    float* dL     = dT   + (size_t)NTOK * DI_;       // NTOK*DI
    float* yact   = dL   + (size_t)NTOK * DI_;       // NTOK*DI
    float* dbc    = yact + (size_t)NTOK * DI_;       // NTOK*56

    const dim3 blk(128);

    // 1) h = gelu(x @ W_in + b_in)            M=2048 K=192 N=384
    gemm_epi_kernel<DM_, DI_, DI_, DM_, DM_, DI_, 1>
        <<<dim3(NTOK / 16, DI_ / 64), blk, 0, stream>>>(x, W_in, b_in, h);

    // 2) dbc = h @ W_x + b_x                  M=2048 K=384 N=56 (pad to 64)
    gemm_epi_kernel<DI_, PC_, PC_, DI_, DI_, PC_, 0>
        <<<dim3(NTOK / 16, 1), blk, 0, stream>>>(h, W_x, b_x, dbc);

    // 3) deltaT = softplus(dbc[:,0:12] @ W_dtT + b_dtT)   K=12 padded to 32
    gemm_epi_kernel<PC_, DI_, DI_, 32, RK_, DI_, 2>
        <<<dim3(NTOK / 16, DI_ / 64), blk, 0, stream>>>(dbc, W_dtT, b_dtT, dT);

    // 4) deltaL = softplus(dbc[:,12:24] @ W_dtL + b_dtL)
    gemm_epi_kernel<PC_, DI_, DI_, 32, RK_, DI_, 2>
        <<<dim3(NTOK / 16, DI_ / 64), blk, 0, stream>>>(dbc + RK_, W_dtL, b_dtL, dL);

    // 5) fused discretize + 2-D scan + C-reduction + h*D + GELU
    scan_kernel<<<dim3((2 * DI_ * DS_) / 256), dim3(256), 0, stream>>>(
        h, dT, dL, dbc, AT_log, AL_log, Dw, yact);

    // 6) out = yact @ W_out + b_out           M=2048 K=384 N=192
    gemm_epi_kernel<DI_, DM_, DM_, DI_, DI_, DM_, 0>
        <<<dim3(NTOK / 16, DM_ / 64), blk, 0, stream>>>(yact, W_out, b_out, out);
}
